// NestedPPGN_40802189312817
// MI455X (gfx1250) — matrix operand order repaired and checked by reading, not compiled, not run on hardware
//
#include <hip/hip_runtime.h>

// ---------------- types for CDNA5 WMMA ----------------
typedef __bf16 bf16_t;
typedef __attribute__((ext_vector_type(16))) __bf16  v16bf;
typedef __attribute__((ext_vector_type(8)))  float   v8f;
typedef __attribute__((ext_vector_type(2)))  float   v2f;
typedef __attribute__((ext_vector_type(4)))  unsigned int u32x4;

#define CH 320          // padded activation channel count (multiple of 32)
#define EMBD 300
#define NTASKS 128

// Swizzled activation layout: buffer dims (Bcount, cchunks=channels/32, ntiles=positions/16).
// Within a (b, kt, nt) tile of 32ch x 16pos (512 elems), lane L of a wave owns 16
// contiguous bf16 (32 bytes): lane = ((k>>4)<<4) | n, element = k&15  (k=c%32, n=p%16).
// Contiguous k order (k = 16*half + element); k_gemm loads its A operand in the same order.
__device__ __forceinline__ size_t swz(int b, int c, int p, int cchunks, int ntiles) {
  int kt = c >> 5, k = c & 31, nt = p >> 4, n = p & 15;
  return ((((size_t)b * cchunks + kt) * ntiles + nt) << 9) + ((((k >> 4) << 4) | n) << 4) + (k & 15);
}

// =====================================================================
// Embedding kernels
// =====================================================================
__global__ void k_node_emb(const int* __restrict__ x, const int* __restrict__ z,
                           const float* __restrict__ z_emb, const float* __restrict__ atom_table,
                           float* __restrict__ node_emb) {
  int idx = blockIdx.x * blockDim.x + threadIdx.x;
  if (idx >= 4096 * 308) return;
  int n = idx / 308, c = idx % 308;
  float v;
  if (c < 8) {
    v = z_emb[z[n] * 8 + c];
  } else {
    const int off[9] = {0, 119, 123, 135, 147, 157, 163, 169, 171};
    int ce = c - 8;
    v = 0.f;
    for (int j = 0; j < 9; ++j) v += atom_table[(x[n * 9 + j] + off[j]) * 300 + ce];
  }
  node_emb[idx] = v;
}

__global__ void k_edge_feat(const int* __restrict__ ea, const float* __restrict__ bond_table,
                            float* __restrict__ ef) {
  int idx = blockIdx.x * blockDim.x + threadIdx.x;
  if (idx >= 16384 * 301) return;
  int e = idx / 301, c = idx % 301;
  float v;
  if (c == 0) {
    v = 1.f;
  } else {
    const int off[3] = {0, 5, 11};
    v = 0.f;
    for (int j = 0; j < 3; ++j) v += bond_table[(ea[e * 3 + j] + off[j]) * 300 + (c - 1)];
  }
  ef[idx] = v;
}

// =====================================================================
// Build h0 (swizzled, dims (256, 20, 16)).  h[b,c,i,j] (p = i*16+j):
//   c<301      : sum of edge features of edges with src_local==j, dst_local==i
//   301<=c<609 : (i==j) ? node_emb[b*16+j][c-301] : 0
// =====================================================================
__global__ void k_h0(const float* __restrict__ ef, const float* __restrict__ node_emb,
                     const int* __restrict__ edge_index, bf16_t* __restrict__ h0) {
  int c = blockIdx.x;   // 0..639
  int b = blockIdx.y;   // 0..255
  int p = threadIdx.x;  // 0..255
  __shared__ int ls[64], ld[64];
  if (threadIdx.x < 64) {
    int e = b * 64 + threadIdx.x;
    ls[threadIdx.x] = edge_index[e] - b * 16;
    ld[threadIdx.x] = edge_index[16384 + e] - b * 16;
  }
  __syncthreads();
  int i = p >> 4, j = p & 15;
  float v = 0.f;
  if (c < 301) {
    for (int e = 0; e < 64; ++e)
      if (ls[e] == j && ld[e] == i) v += ef[(size_t)(b * 64 + e) * 301 + c];
  } else if (c < 609) {
    if (i == j) v = node_emb[(size_t)(b * 16 + j) * 308 + (c - 301)];
  }
  h0[swz(b, c, p, 20, 16)] = (bf16_t)v;
}

// =====================================================================
// Weight packing helpers (f32 -> padded bf16, row-major [Opad][WCpad])
// =====================================================================
__global__ void k_zero_bf16(bf16_t* p, size_t n) {
  size_t i = (size_t)blockIdx.x * blockDim.x + threadIdx.x;
  size_t stride = (size_t)gridDim.x * blockDim.x;
  for (; i < n; i += stride) p[i] = (bf16_t)0.f;
}

__global__ void k_packW(const float* __restrict__ W, int Oreal, int Csrc, int c0, int ccount,
                        bf16_t* __restrict__ dst, int WCpad, int colBase) {
  int idx = blockIdx.x * blockDim.x + threadIdx.x;
  if (idx >= Oreal * ccount) return;
  int o = idx / ccount, cc = idx % ccount;
  dst[(size_t)o * WCpad + colBase + cc] = (bf16_t)W[(size_t)o * Csrc + c0 + cc];
}

// transpose-pack fc inputs [Nrows][Ccols] f32 -> swizzled bf16 (1, Cpad/32, Npad/16)
__global__ void k_packT(const float* __restrict__ src, bf16_t* __restrict__ dst,
                        int Nrows, int Ccols, int Cpad, int Npad) {
  int idx = blockIdx.x * blockDim.x + threadIdx.x;
  if (idx >= Cpad * Npad) return;
  int cc = idx / Npad, n = idx % Npad;
  float v = (cc < Ccols && n < Nrows) ? src[(size_t)n * Ccols + cc] : 0.f;
  dst[swz(0, cc, n, Cpad >> 5, Npad >> 4)] = (bf16_t)v;
}

// =====================================================================
// WMMA GEMM (1x1 conv):  Y[b][o][n] = sum_c W[o][c] * X[b][c][n]
// Wave computes a 32(o) x 64(n) macro-tile: 2x4 register-blocked 16x16 WMMA tiles.
// Requires ntiles % 4 == 0 (all call sites pad to this).
// All B/A accesses: single base pointer + immediate offsets -> global_load_b128.
// =====================================================================
__global__ void __launch_bounds__(128)
k_gemm(const bf16_t* __restrict__ W, int WCpad,
       const bf16_t* __restrict__ X0, int chunks0,
       const bf16_t* __restrict__ X1, int chunks1,
       const float* __restrict__ bias, int Oreal, int relu,
       bf16_t* __restrict__ Y, float* __restrict__ Yf,
       int outChunks, int Nb, int ntiles, int ogroups, int ngroups, int Bcount) {
  int wid = blockIdx.x * (blockDim.x >> 5) + (threadIdx.x >> 5);
  int total = Bcount * ogroups * ngroups;
  if (wid >= total) return;
  int lane = threadIdx.x & 31;
  int b  = wid / (ogroups * ngroups);
  int rm = wid - b * (ogroups * ngroups);
  int og = rm / ngroups, ng = rm - og * ngroups;
  int o0 = og * 32;
  int nt0 = ng * 4;
  int half = lane >> 4, lo = lane & 15;
  int ah = half * 8;

  v8f acc[2][4];
#pragma unroll
  for (int mt = 0; mt < 2; ++mt)
#pragma unroll
    for (int t = 0; t < 4; ++t)
#pragma unroll
      for (int q = 0; q < 8; ++q) acc[mt][t][q] = 0.f;

  const bf16_t* wr0 = W + (size_t)(o0 + lo) * WCpad;        // A rows for o-tile 0
  const bf16_t* wr1 = wr0 + (size_t)16 * WCpad;             // o-tile 1
  size_t step = (size_t)ntiles << 9;

  // ---------------- phase 0 ----------------
  {
    const bf16_t* xp = X0 + (((size_t)b * chunks0 * ntiles + nt0) << 9) + ((size_t)lane << 4);
    for (int ch = 0; ch < chunks0; ++ch) {
      int cb = ch * 32;
      union { u32x4 q[2]; v16bf v; } a0, a1;
      a0.q[0] = *(const u32x4*)(wr0 + cb + 2 * ah);
      a0.q[1] = *(const u32x4*)(wr0 + cb + 2 * ah + 8);
      a1.q[0] = *(const u32x4*)(wr1 + cb + 2 * ah);
      a1.q[1] = *(const u32x4*)(wr1 + cb + 2 * ah + 8);
      union { u32x4 q[2]; v16bf v; } bu[4];
#pragma unroll
      for (int t = 0; t < 4; ++t) {
        bu[t].q[0] = *(const u32x4*)(xp + (t << 9));       // immediate offsets
        bu[t].q[1] = *(const u32x4*)(xp + (t << 9) + 8);
      }
      xp += step;
#pragma unroll
      for (int t = 0; t < 4; ++t) {
        acc[0][t] = __builtin_amdgcn_wmma_f32_16x16x32_bf16(false, a0.v, false, bu[t].v,
                                                            (short)0, acc[0][t], false, false);
        acc[1][t] = __builtin_amdgcn_wmma_f32_16x16x32_bf16(false, a1.v, false, bu[t].v,
                                                            (short)0, acc[1][t], false, false);
      }
    }
  }
  // ---------------- phase 1 (concat operand; W cols at chunks0*32) ----------------
  if (X1) {
    const bf16_t* xp = X1 + (((size_t)b * chunks1 * ntiles + nt0) << 9) + ((size_t)lane << 4);
    int colb = chunks0 * 32;
    for (int ch = 0; ch < chunks1; ++ch) {
      int cb = colb + ch * 32;
      union { u32x4 q[2]; v16bf v; } a0, a1;
      a0.q[0] = *(const u32x4*)(wr0 + cb + 2 * ah);
      a0.q[1] = *(const u32x4*)(wr0 + cb + 2 * ah + 8);
      a1.q[0] = *(const u32x4*)(wr1 + cb + 2 * ah);
      a1.q[1] = *(const u32x4*)(wr1 + cb + 2 * ah + 8);
      union { u32x4 q[2]; v16bf v; } bu[4];
#pragma unroll
      for (int t = 0; t < 4; ++t) {
        bu[t].q[0] = *(const u32x4*)(xp + (t << 9));
        bu[t].q[1] = *(const u32x4*)(xp + (t << 9) + 8);
      }
      xp += step;
#pragma unroll
      for (int t = 0; t < 4; ++t) {
        acc[0][t] = __builtin_amdgcn_wmma_f32_16x16x32_bf16(false, a0.v, false, bu[t].v,
                                                            (short)0, acc[0][t], false, false);
        acc[1][t] = __builtin_amdgcn_wmma_f32_16x16x32_bf16(false, a1.v, false, bu[t].v,
                                                            (short)0, acc[1][t], false, false);
      }
    }
  }
  // ---------------- epilogue ----------------
  // D layout: VGPR q -> M = q + 8*half, N = lo.  A lane's 8 results are 8 consecutive
  // channels at fixed position -> one packed b128 store per tile in the swizzled layout.
  // o0 is 32-aligned: kt = o0>>5 constant; mt toggles khi -> +256 elems; t -> +512 elems.
  bf16_t* sb = Y + ((((size_t)b * outChunks + (o0 >> 5)) * ntiles + nt0) << 9) +
               ((size_t)lo << 4) + ah;
#pragma unroll
  for (int mt = 0; mt < 2; ++mt) {
    int obase = o0 + mt * 16;
#pragma unroll
    for (int t = 0; t < 4; ++t) {
      union { bf16_t h[8]; u32x4 q; } pk;
#pragma unroll
      for (int q = 0; q < 8; ++q) {
        int o = obase + ah + q;
        float v = acc[mt][t][q];
        if (bias && o < Oreal) v += bias[o];
        if (relu && v < 0.f) v = 0.f;
        pk.h[q] = (bf16_t)v;
        if (Yf) Yf[((size_t)b * (outChunks * 32) + o) * Nb + (nt0 + t) * 16 + lo] = v;
      }
      *(u32x4*)(sb + mt * 256 + (t << 9)) = pk.q;
    }
  }
}

// =====================================================================
// Per-(b,c) KxK matmul  mult = M1 @ M2  via V_WMMA_F32_16X16X4_F32.
// Inputs/outputs in swizzled layout (cchunks=10, ntiles=Kdim*Kdim/16).
// =====================================================================
__global__ void __launch_bounds__(128)
k_mult(const bf16_t* __restrict__ M1, const bf16_t* __restrict__ M2,
       bf16_t* __restrict__ Out, int Kdim, int Bcount) {
  int Kt = Kdim >> 4;
  int ntl = (Kdim * Kdim) >> 4;
  int per_b = CH * Kt * Kt;
  int wid = blockIdx.x * (blockDim.x >> 5) + (threadIdx.x >> 5);
  if (wid >= Bcount * per_b) return;
  int lane = threadIdx.x & 31;
  int b = wid / per_b;
  int rr = wid - b * per_b;
  int c = rr / (Kt * Kt); rr -= c * (Kt * Kt);
  int ti = rr / Kt, tj = rr - ti * Kt;
  int m = lane & 15, half = lane >> 4;
  int mg = ti * 16 + m;
  int ng = tj * 16 + m;  // N = lane&15 for B/D
  v8f acc;
#pragma unroll
  for (int q = 0; q < 8; ++q) acc[q] = 0.f;
  for (int kk = 0; kk < Kdim; kk += 4) {
    int ka = kk + half * 2;  // f32 A/B layout: VGPR0 = K{0|2}, VGPR1 = K{1|3}
    v2f a, bb;
    a[0]  = (float)M1[swz(b, c, mg * Kdim + ka, 10, ntl)];
    a[1]  = (float)M1[swz(b, c, mg * Kdim + ka + 1, 10, ntl)];
    bb[0] = (float)M2[swz(b, c, ka * Kdim + ng, 10, ntl)];
    bb[1] = (float)M2[swz(b, c, (ka + 1) * Kdim + ng, 10, ntl)];
    acc = __builtin_amdgcn_wmma_f32_16x16x4_f32(false, a, false, bb,
                                                (short)0, acc, false, false);
  }
#pragma unroll
  for (int q = 0; q < 8; ++q) {
    int mo = ti * 16 + q + half * 8;
    Out[swz(b, c, mo * Kdim + ng, 10, ntl)] = (bf16_t)acc[q];
  }
}

// =====================================================================
// diag_offdiag_meanpool: swizzled [b][CH][K*K] bf16 -> [b][600] f32
// =====================================================================
__global__ void k_pool(const bf16_t* __restrict__ Y, float* __restrict__ out,
                       int Kdim, int Bcount) {
  int idx = blockIdx.x * blockDim.x + threadIdx.x;
  if (idx >= Bcount * EMBD) return;
  int b = idx / EMBD, c = idx % EMBD;
  int ntl = (Kdim * Kdim) >> 4;
  float diag = 0.f, tot = 0.f;
  for (int i = 0; i < Kdim; ++i) diag += (float)Y[swz(b, c, i * Kdim + i, 10, ntl)];
  for (int q = 0; q < Kdim * Kdim; ++q) tot += (float)Y[swz(b, c, q, 10, ntl)];
  out[(size_t)b * (2 * EMBD) + c] = diag / (float)Kdim;
  out[(size_t)b * (2 * EMBD) + EMBD + c] = (tot - diag) / (float)(Kdim * Kdim - Kdim);
}

// =====================================================================
// Build h2 (swizzled, dims (8, 10, 64)), graph level K=32
// =====================================================================
__global__ void k_h2(const bf16_t* __restrict__ hg /* swizzled (1,10,16) */,
                     const int* __restrict__ oei, bf16_t* __restrict__ h2) {
  int c = blockIdx.x;  // 0..319
  int g = blockIdx.y;  // 0..7
  int p = blockIdx.z * 256 + threadIdx.x;  // 0..1023
  __shared__ int ls[128], ld[128];
  if (threadIdx.x < 128) {
    int e = g * 128 + threadIdx.x;
    ls[threadIdx.x] = oei[e] - g * 32;
    ld[threadIdx.x] = oei[1024 + e] - g * 32;
  }
  __syncthreads();
  int i = p >> 5, j = p & 31;
  float v = 0.f;
  if (c == 0) {
    for (int e = 0; e < 128; ++e)
      if (ls[e] == j && ld[e] == i) v += 1.f;
  } else if (c < 301) {
    if (i == j) v = (float)hg[swz(0, c - 1, g * 32 + j, 10, 16)];
  }
  h2[swz(g, c, p, 10, 64)] = (bf16_t)v;
}

__global__ void k_out(const float* __restrict__ Yf, float* __restrict__ out) {
  int idx = blockIdx.x * blockDim.x + threadIdx.x;
  if (idx >= 8 * NTASKS) return;
  int g = idx / NTASKS, t = idx % NTASKS;
  out[idx] = Yf[t * 64 + g];   // fc2 GEMM ran with Nb=64 (padded)
}

// =====================================================================
// Host-side orchestration (all launches on `stream`; capture-safe)
// =====================================================================
static inline int cdiv(int a, int b) { return (a + b - 1) / b; }

struct RBp {
  const float *m1W0, *m1b0, *m1W1, *m1b1, *m2W0, *m2b0, *m2W1, *m2b1, *skW, *skb;
};

static void conv(hipStream_t s, bf16_t* Wbuf,
                 const float* Wsrc, int Oreal, int CsrcTotal,
                 const float* bias, int relu,
                 const bf16_t* X0, int chunks0, int c0real,
                 const bf16_t* X1, int c1real,
                 bf16_t* Y, float* Yf, int outChunks, int otiles,
                 int Nb, int Bcount) {
  int chunks1 = X1 ? (CH / 32) : 0;
  int WCpad = (chunks0 + chunks1) * 32;
  int Opad = otiles * 16;
  size_t wn = (size_t)Opad * WCpad;
  k_zero_bf16<<<256, 256, 0, s>>>(Wbuf, wn);
  k_packW<<<cdiv(Oreal * c0real, 256), 256, 0, s>>>(Wsrc, Oreal, CsrcTotal, 0, c0real,
                                                    Wbuf, WCpad, 0);
  if (X1)
    k_packW<<<cdiv(Oreal * c1real, 256), 256, 0, s>>>(Wsrc, Oreal, CsrcTotal, c0real, c1real,
                                                      Wbuf, WCpad, chunks0 * 32);
  int ntiles = Nb / 16;        // must be a multiple of 4 (all call sites)
  int ngroups = ntiles / 4;
  int ogroups = otiles / 2;
  int waves = Bcount * ogroups * ngroups;
  k_gemm<<<cdiv(waves, 4), 128, 0, s>>>(Wbuf, WCpad, X0, chunks0, X1, chunks1,
                                        bias, Oreal, relu, Y, Yf, outChunks, Nb, ntiles,
                                        ogroups, ngroups, Bcount);
}

static void run_block(hipStream_t s, const RBp& p, int cin,
                      const bf16_t* X0, int x0chunks,
                      bf16_t* Xa, bf16_t* Tm1, bf16_t* Tm2, bf16_t* Xout,
                      bf16_t* Wbuf, int Bcount, int Nb, int Kdim) {
  // m1 = relu(conv(relu(conv(X0))))
  conv(s, Wbuf, p.m1W0, EMBD, cin, p.m1b0, 1, X0, x0chunks, cin, nullptr, 0,
       Xa, nullptr, 10, 20, Nb, Bcount);
  conv(s, Wbuf, p.m1W1, EMBD, EMBD, p.m1b1, 1, Xa, 10, EMBD, nullptr, 0,
       Tm1, nullptr, 10, 20, Nb, Bcount);
  // m2
  conv(s, Wbuf, p.m2W0, EMBD, cin, p.m2b0, 1, X0, x0chunks, cin, nullptr, 0,
       Xa, nullptr, 10, 20, Nb, Bcount);
  conv(s, Wbuf, p.m2W1, EMBD, EMBD, p.m2b1, 1, Xa, 10, EMBD, nullptr, 0,
       Tm2, nullptr, 10, 20, Nb, Bcount);
  // mult = per-(b,c) KxK matmul  (reuse Xa as output)
  {
    int Kt = Kdim >> 4;
    int waves = Bcount * CH * Kt * Kt;
    k_mult<<<cdiv(waves, 4), 128, 0, s>>>(Tm1, Tm2, Xa, Kdim, Bcount);
  }
  // skip = conv1x1(concat(X0, mult))  (no relu)
  conv(s, Wbuf, p.skW, EMBD, cin + EMBD, p.skb, 0, X0, x0chunks, cin, Xa, EMBD,
       Xout, nullptr, 10, 20, Nb, Bcount);
}

extern "C" void kernel_launch(void* const* d_in, const int* in_sizes, int n_in,
                              void* d_out, int out_size, void* d_ws, size_t ws_size,
                              hipStream_t stream) {
  (void)in_sizes; (void)n_in; (void)out_size; (void)ws_size;
  auto f  = [&](int i) { return (const float*)d_in[i]; };
  auto ii = [&](int i) { return (const int*)d_in[i]; };

  // ---- input mapping (setup_inputs dict order, depth-first) ----
  int k = 0;
  const float* z_embT = f(k++);   // params.z_emb [1000,8]
  const float* atomT  = f(k++);   // params.atom_table [173,300]
  const float* bondT  = f(k++);   // params.bond_table [13,300]
  RBp rb1[4], rb2[4];
  for (int b = 0; b < 4; ++b) {
    rb1[b] = {f(k), f(k + 1), f(k + 2), f(k + 3), f(k + 4), f(k + 5), f(k + 6), f(k + 7),
              f(k + 8), f(k + 9)};
    k += 10;
  }
  const float* fcgW0 = f(k++); const float* fcgb0 = f(k++);
  const float* fcgW1 = f(k++); const float* fcgb1 = f(k++);
  for (int b = 0; b < 4; ++b) {
    rb2[b] = {f(k), f(k + 1), f(k + 2), f(k + 3), f(k + 4), f(k + 5), f(k + 6), f(k + 7),
              f(k + 8), f(k + 9)};
    k += 10;
  }
  const float* fcW0 = f(k++); const float* fcb0 = f(k++);
  const float* fcW1 = f(k++); const float* fcb1 = f(k++);
  const int* x_in      = ii(k++);  // [4096,9]
  const int* ea_in     = ii(k++);  // [16384,3]
  const int* z_in      = ii(k++);  // [4096]
  const int* eidx_in   = ii(k++);  // [2,16384]
  k++;                             // node_to_subgraph (implicit: n/16)
  k++;                             // subgraph_to_graph (implicit: s/32)
  const int* oei_in    = ii(k++);  // [2,1024]

  // ---- workspace bump allocator ----
  size_t off = 0;
  auto alloc = [&](size_t bytes) {
    void* p = (char*)d_ws + off;
    off += (bytes + 255) & ~(size_t)255;
    return p;
  };
  float*  node_emb = (float*)alloc((size_t)4096 * 308 * 4);
  float*  ef       = (float*)alloc((size_t)16384 * 301 * 4);
  bf16_t* h0   = (bf16_t*)alloc((size_t)256 * 640 * 256 * 2);
  bf16_t* Xp   = (bf16_t*)alloc((size_t)256 * CH * 256 * 2);
  bf16_t* Xq   = (bf16_t*)alloc((size_t)256 * CH * 256 * 2);
  bf16_t* Xa   = (bf16_t*)alloc((size_t)256 * CH * 256 * 2);
  bf16_t* Tm1  = (bf16_t*)alloc((size_t)256 * CH * 256 * 2);
  bf16_t* Tm2  = (bf16_t*)alloc((size_t)256 * CH * 256 * 2);
  bf16_t* Wbuf = (bf16_t*)alloc((size_t)320 * 960 * 2);
  float*  pool1 = (float*)alloc((size_t)256 * 600 * 4);
  bf16_t* fcX  = (bf16_t*)alloc((size_t)640 * 256 * 2);
  bf16_t* fcA  = (bf16_t*)alloc((size_t)CH * 256 * 2);
  bf16_t* fcB  = (bf16_t*)alloc((size_t)CH * 256 * 2);
  bf16_t* h2   = (bf16_t*)alloc((size_t)8 * CH * 1024 * 2);
  bf16_t* X2p  = (bf16_t*)alloc((size_t)8 * CH * 1024 * 2);
  bf16_t* X2q  = (bf16_t*)alloc((size_t)8 * CH * 1024 * 2);
  bf16_t* X2a  = (bf16_t*)alloc((size_t)8 * CH * 1024 * 2);
  bf16_t* T2m1 = (bf16_t*)alloc((size_t)8 * CH * 1024 * 2);
  bf16_t* T2m2 = (bf16_t*)alloc((size_t)8 * CH * 1024 * 2);
  float*  pool2 = (float*)alloc((size_t)8 * 600 * 4);
  bf16_t* fc2X = (bf16_t*)alloc((size_t)640 * 64 * 2);   // Nb padded to 64
  bf16_t* fc2A = (bf16_t*)alloc((size_t)CH * 64 * 2);
  bf16_t* fc2B = (bf16_t*)alloc((size_t)128 * 64 * 2);
  float*  fc2Yf = (float*)alloc((size_t)128 * 64 * 4);

  hipStream_t s = stream;

  // ---- stage A: embeddings & h0 ----
  k_node_emb<<<cdiv(4096 * 308, 256), 256, 0, s>>>(x_in, z_in, z_embT, atomT, node_emb);
  k_edge_feat<<<cdiv(16384 * 301, 256), 256, 0, s>>>(ea_in, bondT, ef);
  k_h0<<<dim3(640, 256), 256, 0, s>>>(ef, node_emb, eidx_in, h0);

  // ---- stage B: rb1 x4 (B=256, K=16, Nb=256) ----
  run_block(s, rb1[0], 609, h0, 20, Xa, Tm1, Tm2, Xp, Wbuf, 256, 256, 16);
  run_block(s, rb1[1], 300, Xp, 10, Xa, Tm1, Tm2, Xq, Wbuf, 256, 256, 16);
  run_block(s, rb1[2], 300, Xq, 10, Xa, Tm1, Tm2, Xp, Wbuf, 256, 256, 16);
  run_block(s, rb1[3], 300, Xp, 10, Xa, Tm1, Tm2, Xq, Wbuf, 256, 256, 16);

  // ---- pooling + fc_g ----
  k_pool<<<cdiv(256 * EMBD, 256), 256, 0, s>>>(Xq, pool1, 16, 256);
  k_packT<<<cdiv(640 * 256, 256), 256, 0, s>>>(pool1, fcX, 256, 600, 640, 256);
  conv(s, Wbuf, fcgW0, EMBD, 600, fcgb0, 1, fcX, 20, 600, nullptr, 0,
       fcA, nullptr, 10, 20, 256, 1);
  conv(s, Wbuf, fcgW1, EMBD, EMBD, fcgb1, 1, fcA, 10, EMBD, nullptr, 0,
       fcB, nullptr, 10, 20, 256, 1);

  // ---- stage C: graph level (B=8, K=32, Nb=1024) ----
  k_h2<<<dim3(CH, 8, 4), 256, 0, s>>>(fcB, oei_in, h2);
  run_block(s, rb2[0], 301, h2, 10, X2a, T2m1, T2m2, X2p, Wbuf, 8, 1024, 32);
  run_block(s, rb2[1], 300, X2p, 10, X2a, T2m1, T2m2, X2q, Wbuf, 8, 1024, 32);
  run_block(s, rb2[2], 300, X2q, 10, X2a, T2m1, T2m2, X2p, Wbuf, 8, 1024, 32);
  run_block(s, rb2[3], 300, X2p, 10, X2a, T2m1, T2m2, X2q, Wbuf, 8, 1024, 32);

  k_pool<<<cdiv(8 * EMBD, 256), 256, 0, s>>>(X2q, pool2, 32, 8);
  k_packT<<<cdiv(640 * 64, 256), 256, 0, s>>>(pool2, fc2X, 8, 600, 640, 64);
  conv(s, Wbuf, fcW0, EMBD, 600, fcb0, 1, fc2X, 20, 600, nullptr, 0,
       fc2A, nullptr, 10, 20, 64, 1);
  conv(s, Wbuf, fcW1, NTASKS, EMBD, fcb1, 0, fc2A, 10, EMBD, nullptr, 0,
       fc2B, fc2Yf, 4, 8, 64, 1);
  k_out<<<cdiv(8 * NTASKS, 256), 256, 0, s>>>(fc2Yf, (float*)d_out);
}
